// VisionTransformer_77876347011239
// MI455X (gfx1250) — matrix operand order repaired and checked by reading, not compiled, not run on hardware
//
#include <hip/hip_runtime.h>
#include <hip/hip_bf16.h>
#include <math.h>

// ---------------------------------------------------------------------------
// ViT forward for MI455X (gfx1250, wave32, WMMA).
// All big matmuls run on v_wmma_f32_16x16x32_f16 (f16 inputs, f32 accum).
// ---------------------------------------------------------------------------

typedef __attribute__((ext_vector_type(16))) _Float16 v16h;
typedef __attribute__((ext_vector_type(8)))  _Float16 h8;
typedef __attribute__((ext_vector_type(8)))  float    v8f;

#define DEPTH_ 12
#define E_     768
#define NH_    12
#define DH_    64
#define NTOK_  49
#define MLP_   3072
#define CLS_   1000
#define B_     128
#define MTOK_  (B_ * NTOK_)   // 6272
#define BIASP_ 4096           // padded 64x64 bias tile per (layer, head)

// -------------------------- small device helpers ---------------------------

__device__ __forceinline__ v8f zero_v8f() {
  v8f z;
#pragma unroll
  for (int i = 0; i < 8; ++i) z[i] = 0.0f;
  return z;
}

__device__ __forceinline__ h8 zero_h8() {
  h8 z;
#pragma unroll
  for (int i = 0; i < 8; ++i) z[i] = (_Float16)0;
  return z;
}

__device__ __forceinline__ v8f wmma16(v16h a, v16h b, v8f c) {
  // D = A(16x32 f16) * B(32x16 f16) + C(16x16 f32)
  return __builtin_amdgcn_wmma_f32_16x16x32_f16(
      /*neg_a=*/false, a, /*neg_b=*/false, b,
      /*c_mod=*/(short)0, c, /*reuse_a=*/false, /*reuse_b=*/false);
}

// A-fragment (16x32, f16) from LDS row-major tile, stride lda halfs.
// Lane L: m = L&15, half = L>>4. Element j: k = (j<8 ? half*8+j : 16+half*8+j-8)
// -> two contiguous 8-half (16B) runs at k0+half*8 and k0+16+half*8.
__device__ __forceinline__ v16h frag_a(const _Float16* base, int lda, int row0,
                                       int k0, int lane) {
  int m = lane & 15, hf = lane >> 4;
  const _Float16* p = base + (row0 + m) * lda + k0 + hf * 8;
  h8 lo = *(const h8*)p;
  h8 hi = *(const h8*)(p + 16);
  v16h a;
#pragma unroll
  for (int i = 0; i < 8; ++i) { a[i] = lo[i]; a[8 + i] = hi[i]; }
  return a;
}

// B-fragment (32x16, f16). LDS tile stored n-major: Bs[n][k], stride ldb.
// Lane L: n = L&15, half = L>>4. Element j: k = (j<8 ? half*8+j : 16+half*8+j-8)
// -> two contiguous 8-half (16B) runs at k0+half*8 and k0+16+half*8.
__device__ __forceinline__ v16h frag_b(const _Float16* base, int ldb, int col0,
                                       int k0, int lane) {
  int n = lane & 15, hf = lane >> 4;
  const _Float16* p = base + (col0 + n) * ldb + k0 + hf * 8;
  h8 lo = *(const h8*)p;
  h8 hi = *(const h8*)(p + 16);
  v16h b;
#pragma unroll
  for (int i = 0; i < 8; ++i) { b[i] = lo[i]; b[8 + i] = hi[i]; }
  return b;
}

// ------------------------------ GEMM kernel --------------------------------
// C[M,N] = A[M,K](f16) @ W[N,K]^T(f16)  (+ epilogue)
// 256 threads = 8 waves (2 M x 4 N). Workgroup tile 128x128, K-step 32,
// double-buffered LDS so staging tile k+1 overlaps the 8 WMMAs of tile k,
// plus global_prefetch of tile k+2.
// EPI: 0 = f32 out + bias                       (head)
//      1 = f32 out + bias + pos_embed[m%49]     (patch embed)
//      2 = f32 out + bias + residual            (proj / fc2)
//      3 = f16 out, no bias                     (qkv)
//      4 = f16 out, gelu(acc+bias)              (fc1)
// GUARD=false when M,N are multiples of 128 (all but the head GEMM).

template <int EPI, bool GUARD>
__global__ __launch_bounds__(256) void gemm_wmma(
    const _Float16* __restrict__ A, const _Float16* __restrict__ W,
    const float* __restrict__ bias, const float* __restrict__ res,
    const float* __restrict__ pos, float* __restrict__ outF,
    _Float16* __restrict__ outH, int M, int N, int K) {
  __shared__ _Float16 Ash[2][128 * 40];  // stride 40 halfs, rows 16B aligned
  __shared__ _Float16 Bsh[2][128 * 40];  // n-major: Bsh[n][k] = W[tileN+n][k]

  const int tid = threadIdx.x;
  const int lane = tid & 31;
  const int wid = tid >> 5;
  const int waveM = wid & 1;   // 0..1
  const int waveN = wid >> 1;  // 0..3
  const int tileM = blockIdx.y * 128;
  const int tileN = blockIdx.x * 128;
  const int KT = K >> 5;

  // Per-thread staging chunk coordinates (2 chunks of 16B per tile each).
  const int row0 = tid >> 2;            // 0..63
  const int row1 = row0 + 64;           // 64..127
  const int cc = (tid & 3) * 8;

  auto stage = [&](int kt, int buf) {
    const int k0 = kt << 5;
    h8 a0, a1, w0, w1;
    if (GUARD) {
      a0 = (tileM + row0 < M) ? *(const h8*)(A + (size_t)(tileM + row0) * K + k0 + cc) : zero_h8();
      a1 = (tileM + row1 < M) ? *(const h8*)(A + (size_t)(tileM + row1) * K + k0 + cc) : zero_h8();
      w0 = (tileN + row0 < N) ? *(const h8*)(W + (size_t)(tileN + row0) * K + k0 + cc) : zero_h8();
      w1 = (tileN + row1 < N) ? *(const h8*)(W + (size_t)(tileN + row1) * K + k0 + cc) : zero_h8();
    } else {
      a0 = *(const h8*)(A + (size_t)(tileM + row0) * K + k0 + cc);
      a1 = *(const h8*)(A + (size_t)(tileM + row1) * K + k0 + cc);
      w0 = *(const h8*)(W + (size_t)(tileN + row0) * K + k0 + cc);
      w1 = *(const h8*)(W + (size_t)(tileN + row1) * K + k0 + cc);
    }
    *(h8*)(Ash[buf] + row0 * 40 + cc) = a0;
    *(h8*)(Ash[buf] + row1 * 40 + cc) = a1;
    *(h8*)(Bsh[buf] + row0 * 40 + cc) = w0;
    *(h8*)(Bsh[buf] + row1 * 40 + cc) = w1;
    // Pull tile k+2 toward the WGP while WMMAs drain.
    const int kp = k0 + 64;
    if (kp < K) {
      __builtin_prefetch((const void*)(A + (size_t)(tileM + row0) * K + kp + cc), 0, 1);
      __builtin_prefetch((const void*)(W + (size_t)(tileN + row0) * K + kp + cc), 0, 1);
    }
  };

  v8f acc[4][2];
#pragma unroll
  for (int i = 0; i < 4; ++i)
#pragma unroll
    for (int j = 0; j < 2; ++j) acc[i][j] = zero_v8f();

  stage(0, 0);
  for (int kt = 0; kt < KT; ++kt) {
    const int buf = kt & 1;
    __syncthreads();  // buf now fully staged; other buf free for writes
    if (kt + 1 < KT) stage(kt + 1, buf ^ 1);

    v16h bfr[2];
#pragma unroll
    for (int tn = 0; tn < 2; ++tn)
      bfr[tn] = frag_b(Bsh[buf], 40, waveN * 32 + tn * 16, 0, lane);
#pragma unroll
    for (int tm = 0; tm < 4; ++tm) {
      v16h afr = frag_a(Ash[buf], 40, waveM * 64 + tm * 16, 0, lane);
#pragma unroll
      for (int tn = 0; tn < 2; ++tn)
        acc[tm][tn] = wmma16(afr, bfr[tn], acc[tm][tn]);
    }
  }

  // Epilogue. C/D layout: lane n = L&15; VGPR r -> m = r + 8*(L>>4).
  const int nlo = lane & 15;
  const int hf = lane >> 4;
#pragma unroll
  for (int tm = 0; tm < 4; ++tm) {
#pragma unroll
    for (int tn = 0; tn < 2; ++tn) {
      int gn = tileN + waveN * 32 + tn * 16 + nlo;
#pragma unroll
      for (int r = 0; r < 8; ++r) {
        int gm = tileM + waveM * 64 + tm * 16 + hf * 8 + r;
        if (!GUARD || (gm < M && gn < N)) {
          size_t idx = (size_t)gm * N + gn;
          float v = acc[tm][tn][r];
          if constexpr (EPI == 0) {
            outF[idx] = v + bias[gn];
          } else if constexpr (EPI == 1) {
            outF[idx] = v + bias[gn] + pos[(size_t)(gm % NTOK_) * N + gn];
          } else if constexpr (EPI == 2) {
            outF[idx] = v + bias[gn] + res[idx];
          } else if constexpr (EPI == 3) {
            outH[idx] = (_Float16)v;
          } else if constexpr (EPI == 4) {
            v += bias[gn];
            v = 0.5f * v * (1.0f + erff(v * 0.70710678118654752f));  // exact gelu
            outH[idx] = (_Float16)v;
          }
        }
      }
    }
  }
}

// ----------------------------- LayerNorm -----------------------------------
// One block per row (768 cols, 256 threads x 3). f32 stats, f16 output.
__global__ __launch_bounds__(256) void layernorm_f16(
    const float* __restrict__ x, const float* __restrict__ g,
    const float* __restrict__ b, _Float16* __restrict__ out) {
  __shared__ float red[256];
  const int row = blockIdx.x, tid = threadIdx.x;
  const float* xr = x + (size_t)row * E_;
  float v0 = xr[tid], v1 = xr[tid + 256], v2 = xr[tid + 512];
  red[tid] = v0 + v1 + v2;
  __syncthreads();
  for (int off = 128; off; off >>= 1) {
    if (tid < off) red[tid] += red[tid + off];
    __syncthreads();
  }
  float mean = red[0] * (1.0f / E_);
  __syncthreads();
  float d0 = v0 - mean, d1 = v1 - mean, d2 = v2 - mean;
  red[tid] = d0 * d0 + d1 * d1 + d2 * d2;
  __syncthreads();
  for (int off = 128; off; off >>= 1) {
    if (tid < off) red[tid] += red[tid + off];
    __syncthreads();
  }
  float rstd = rsqrtf(red[0] * (1.0f / E_) + 1e-5f);
  _Float16* o = out + (size_t)row * E_;
  o[tid]       = (_Float16)(d0 * rstd * g[tid]       + b[tid]);
  o[tid + 256] = (_Float16)(d1 * rstd * g[tid + 256] + b[tid + 256]);
  o[tid + 512] = (_Float16)(d2 * rstd * g[tid + 512] + b[tid + 512]);
}

// ----------------------------- Attention -----------------------------------
// One workgroup (128 threads = 4 waves) per (b, h). N=49 padded to 64, DH=64.
// S = Q K^T * scale + bias  -> softmax -> O = P V, all matmuls via WMMA.
// biasL is pre-expanded & zero-padded to [H][64][64] so the S epilogue adds it
// unconditionally (no exec-mask branching, coalesced reads).
__global__ __launch_bounds__(128) void attention_kernel(
    const _Float16* __restrict__ qkv,   // [6272, 2304] f16
    const float* __restrict__ biasL,    // [H, 64, 64] f32, zero padded
    _Float16* __restrict__ o) {         // [6272, 768] f16
  __shared__ _Float16 Qsh[64 * 72];
  __shared__ _Float16 Ksh[64 * 72];
  __shared__ _Float16 Vsh[64 * 72];  // transposed: Vsh[d][j] = V[j][d]
  __shared__ _Float16 Psh[64 * 72];
  __shared__ float    Ssh[64 * 68];

  const int bh = blockIdx.x;
  const int b = bh / NH_, h = bh % NH_;
  const int tid = threadIdx.x, lane = tid & 31, wid = tid >> 5;
  const float* bias = biasL + (size_t)h * BIASP_;

  // Load Q, K (row-major, zero-padded rows 49..63); zero V.
  for (int idx = tid; idx < 64 * 8; idx += 128) {
    int row = idx >> 3, cc = (idx & 7) * 8;
    h8 q = zero_h8(), k = zero_h8();
    if (row < NTOK_) {
      const _Float16* base =
          qkv + ((size_t)(b * NTOK_ + row)) * (3 * E_) + h * DH_ + cc;
      q = *(const h8*)base;
      k = *(const h8*)(base + E_);
    }
    *(h8*)(Qsh + row * 72 + cc) = q;
    *(h8*)(Ksh + row * 72 + cc) = k;
  }
  for (int idx = tid; idx < 64 * 72; idx += 128) Vsh[idx] = (_Float16)0;
  __syncthreads();
  for (int idx = tid; idx < NTOK_ * DH_; idx += 128) {
    int j = idx >> 6, d = idx & 63;
    Vsh[d * 72 + j] =
        qkv[((size_t)(b * NTOK_ + j)) * (3 * E_) + 2 * E_ + h * DH_ + d];
  }
  __syncthreads();

  const int nlo = lane & 15, hf = lane >> 4;

  // ---- S = Q K^T (64x64, 16 tiles, one tile-row per wave) ----
  v8f sacc[4];
#pragma unroll
  for (int tn = 0; tn < 4; ++tn) sacc[tn] = zero_v8f();
  for (int kk = 0; kk < DH_; kk += 32) {
    v16h aq = frag_a(Qsh, 72, wid * 16, kk, lane);
#pragma unroll
    for (int tn = 0; tn < 4; ++tn) {
      v16h bk = frag_b(Ksh, 72, tn * 16, kk, lane);
      sacc[tn] = wmma16(aq, bk, sacc[tn]);
    }
  }
#pragma unroll
  for (int tn = 0; tn < 4; ++tn) {
#pragma unroll
    for (int r = 0; r < 8; ++r) {
      int i = wid * 16 + hf * 8 + r;
      int j = tn * 16 + nlo;
      Ssh[i * 68 + j] = sacc[tn][r] * 0.125f + bias[i * 64 + j];  // DH^-0.5
    }
  }
  __syncthreads();

  // ---- softmax over 49 valid columns ----
  if (tid < 64) {
    float* srow = Ssh + tid * 68;
    float mx = -1e30f;
    for (int j = 0; j < NTOK_; ++j) mx = fmaxf(mx, srow[j]);
    float sum = 0.0f;
    for (int j = 0; j < NTOK_; ++j) {
      float e = __expf(srow[j] - mx);
      sum += e;
      srow[j] = e;
    }
    float inv = 1.0f / sum;
    _Float16* prow = Psh + tid * 72;
    for (int j = 0; j < NTOK_; ++j) prow[j] = (_Float16)(srow[j] * inv);
    for (int j = NTOK_; j < 64; ++j) prow[j] = (_Float16)0;
  }
  __syncthreads();

  // ---- O = P V (64x64 over padded tokens) ----
  v8f oacc[4];
#pragma unroll
  for (int tn = 0; tn < 4; ++tn) oacc[tn] = zero_v8f();
  for (int kk = 0; kk < 64; kk += 32) {
    v16h ap = frag_a(Psh, 72, wid * 16, kk, lane);
#pragma unroll
    for (int tn = 0; tn < 4; ++tn) {
      v16h bv = frag_b(Vsh, 72, tn * 16, kk, lane);
      oacc[tn] = wmma16(ap, bv, oacc[tn]);
    }
  }
#pragma unroll
  for (int tn = 0; tn < 4; ++tn) {
#pragma unroll
    for (int r = 0; r < 8; ++r) {
      int i = wid * 16 + hf * 8 + r;
      if (i < NTOK_) {
        int d = tn * 16 + nlo;
        o[((size_t)(b * NTOK_ + i)) * E_ + h * DH_ + d] = (_Float16)oacc[tn][r];
      }
    }
  }
}

// --------------------------- misc small kernels ----------------------------

__global__ void f32_to_f16(const float* __restrict__ s,
                           _Float16* __restrict__ d, int count) {
  int i = blockIdx.x * blockDim.x + threadIdx.x;
  if (i < count) d[i] = (_Float16)s[i];
}

// im2col for the stride-16 16x16 patch conv: out[m, c*256+p*16+q]
__global__ void im2col_patch(const float* __restrict__ x,
                             _Float16* __restrict__ out) {
  int i = blockIdx.x * blockDim.x + threadIdx.x;
  if (i >= MTOK_ * E_) return;
  int m = i / E_, k = i % E_;
  int b = m / NTOK_, n = m % NTOK_;
  int c = k / 256, pq = k % 256, p = pq / 16, q = pq % 16;
  int gh = n / 7, gw = n % 7;
  out[i] = (_Float16)x[(((size_t)b * 3 + c) * 112 + gh * 16 + p) * 112 +
                       gw * 16 + q];
}

// bias_full[d][h][i][j] (i,j in [0,64), zero padded) = rel_bias[d, idx(i,j), h]
__global__ void build_bias(const float* __restrict__ rel,
                           float* __restrict__ out) {
  int i = blockIdx.x * blockDim.x + threadIdx.x;
  const int total = DEPTH_ * NH_ * BIASP_;
  if (i >= total) return;
  int d = i / (NH_ * BIASP_);
  int r = i % (NH_ * BIASP_);
  int h = r / BIASP_;
  int ij = r % BIASP_;
  int ii = ij >> 6, jj = ij & 63;
  float v = 0.0f;
  if (ii < NTOK_ && jj < NTOK_) {
    int r0 = ii / 7 - jj / 7 + 6;
    int r1 = ii % 7 - jj % 7 + 6;
    int idx = r0 * 13 + r1;  // (2G-1)=13
    v = rel[((size_t)d * 169 + idx) * NH_ + h];
  }
  out[i] = v;
}

// mean over 49 tokens of the final-LN f16 activations -> pooled f16 [128,768]
__global__ void meanpool(const _Float16* __restrict__ x,
                         _Float16* __restrict__ out) {
  int i = blockIdx.x * blockDim.x + threadIdx.x;
  if (i >= B_ * E_) return;
  int b = i / E_, e = i % E_;
  float s = 0.0f;
  for (int n = 0; n < NTOK_; ++n)
    s += (float)x[((size_t)(b * NTOK_ + n)) * E_ + e];
  out[i] = (_Float16)(s * (1.0f / NTOK_));
}

// ------------------------------- launcher ----------------------------------

static inline int cdiv_l(long a, long b) { return (int)((a + b - 1) / b); }

extern "C" void kernel_launch(void* const* d_in, const int* in_sizes, int n_in,
                              void* d_out, int out_size, void* d_ws,
                              size_t ws_size, hipStream_t stream) {
  const float* x        = (const float*)d_in[0];
  const float* patch_w  = (const float*)d_in[1];
  const float* patch_b  = (const float*)d_in[2];
  const float* pos_emb  = (const float*)d_in[3];
  const float* ln1_g    = (const float*)d_in[4];
  const float* ln1_b    = (const float*)d_in[5];
  const float* qkv_w    = (const float*)d_in[6];
  const float* rel_bias = (const float*)d_in[7];
  const float* proj_w   = (const float*)d_in[8];
  const float* proj_b   = (const float*)d_in[9];
  const float* ln2_g    = (const float*)d_in[10];
  const float* ln2_b    = (const float*)d_in[11];
  const float* fc1_w    = (const float*)d_in[12];
  const float* fc1_b    = (const float*)d_in[13];
  const float* fc2_w    = (const float*)d_in[14];
  const float* fc2_b    = (const float*)d_in[15];
  const float* norm_g   = (const float*)d_in[16];
  const float* norm_b   = (const float*)d_in[17];
  const float* head_w   = (const float*)d_in[18];
  const float* head_b   = (const float*)d_in[19];
  float* out = (float*)d_out;

  // ---- workspace carving (256B aligned) ----
  char* ws = (char*)d_ws;
  auto alloc = [&](size_t bytes) -> void* {
    void* p = (void*)ws;
    ws += (bytes + 255) & ~(size_t)255;
    return p;
  };
  const size_t nQKV = (size_t)DEPTH_ * 3 * E_ * E_;    // 21.2M
  const size_t nPRJ = (size_t)DEPTH_ * E_ * E_;        // 7.1M
  const size_t nFC  = (size_t)DEPTH_ * MLP_ * E_;      // 28.3M
  _Float16* wq  = (_Float16*)alloc(nQKV * 2);
  _Float16* wp  = (_Float16*)alloc(nPRJ * 2);
  _Float16* w1  = (_Float16*)alloc(nFC * 2);
  _Float16* w2  = (_Float16*)alloc(nFC * 2);
  _Float16* wh  = (_Float16*)alloc((size_t)CLS_ * E_ * 2);
  _Float16* wpa = (_Float16*)alloc((size_t)E_ * E_ * 2);
  _Float16* acol  = (_Float16*)alloc((size_t)MTOK_ * E_ * 2);
  float*    hbuf  = (float*)alloc((size_t)MTOK_ * E_ * 4);
  _Float16* hn    = (_Float16*)alloc((size_t)MTOK_ * E_ * 2);
  _Float16* qkvb  = (_Float16*)alloc((size_t)MTOK_ * 3 * E_ * 2);
  _Float16* obuf  = (_Float16*)alloc((size_t)MTOK_ * E_ * 2);
  _Float16* mlpb  = (_Float16*)alloc((size_t)MTOK_ * MLP_ * 2);
  float*    biasf = (float*)alloc((size_t)DEPTH_ * NH_ * BIASP_ * 4);
  _Float16* lnf   = (_Float16*)alloc((size_t)MTOK_ * E_ * 2);
  _Float16* pooled = (_Float16*)alloc((size_t)B_ * E_ * 2);

  // ---- weight conversion f32 -> f16 ----
  f32_to_f16<<<cdiv_l((long)nQKV, 256), 256, 0, stream>>>(qkv_w, wq, (int)nQKV);
  f32_to_f16<<<cdiv_l((long)nPRJ, 256), 256, 0, stream>>>(proj_w, wp, (int)nPRJ);
  f32_to_f16<<<cdiv_l((long)nFC, 256), 256, 0, stream>>>(fc1_w, w1, (int)nFC);
  f32_to_f16<<<cdiv_l((long)nFC, 256), 256, 0, stream>>>(fc2_w, w2, (int)nFC);
  f32_to_f16<<<cdiv_l((long)CLS_ * E_, 256), 256, 0, stream>>>(head_w, wh,
                                                              CLS_ * E_);
  f32_to_f16<<<cdiv_l((long)E_ * E_, 256), 256, 0, stream>>>(patch_w, wpa,
                                                             E_ * E_);
  build_bias<<<cdiv_l((long)DEPTH_ * NH_ * BIASP_, 256), 256, 0, stream>>>(
      rel_bias, biasf);

  // ---- patch embed: im2col + GEMM(+bias+pos) -> h (f32) ----
  im2col_patch<<<cdiv_l((long)MTOK_ * E_, 256), 256, 0, stream>>>(x, acol);
  {
    dim3 g(E_ / 128, MTOK_ / 128);
    gemm_wmma<1, false><<<g, 256, 0, stream>>>(acol, wpa, patch_b, nullptr,
                                               pos_emb, hbuf, nullptr, MTOK_,
                                               E_, E_);
  }

  // ---- transformer blocks ----
  for (int d = 0; d < DEPTH_; ++d) {
    layernorm_f16<<<MTOK_, 256, 0, stream>>>(hbuf, ln1_g + d * E_,
                                             ln1_b + d * E_, hn);
    {  // qkv: [6272,768] x [2304,768]^T -> f16, no bias
      dim3 g((3 * E_) / 128, MTOK_ / 128);
      gemm_wmma<3, false><<<g, 256, 0, stream>>>(
          hn, wq + (size_t)d * 3 * E_ * E_, nullptr, nullptr, nullptr, nullptr,
          qkvb, MTOK_, 3 * E_, E_);
    }
    attention_kernel<<<B_ * NH_, 128, 0, stream>>>(
        qkvb, biasf + (size_t)d * NH_ * BIASP_, obuf);
    {  // proj + residual -> h (f32)
      dim3 g(E_ / 128, MTOK_ / 128);
      gemm_wmma<2, false><<<g, 256, 0, stream>>>(
          obuf, wp + (size_t)d * E_ * E_, proj_b + d * E_, hbuf, nullptr, hbuf,
          nullptr, MTOK_, E_, E_);
    }
    layernorm_f16<<<MTOK_, 256, 0, stream>>>(hbuf, ln2_g + d * E_,
                                             ln2_b + d * E_, hn);
    {  // fc1 + gelu -> f16
      dim3 g(MLP_ / 128, MTOK_ / 128);
      gemm_wmma<4, false><<<g, 256, 0, stream>>>(
          hn, w1 + (size_t)d * MLP_ * E_, fc1_b + d * MLP_, nullptr, nullptr,
          nullptr, mlpb, MTOK_, MLP_, E_);
    }
    {  // fc2 + residual -> h (f32)
      dim3 g(E_ / 128, MTOK_ / 128);
      gemm_wmma<2, false><<<g, 256, 0, stream>>>(
          mlpb, w2 + (size_t)d * E_ * MLP_, fc2_b + d * E_, hbuf, nullptr,
          hbuf, nullptr, MTOK_, E_, MLP_);
    }
  }

  // ---- final LN -> mean pool -> head ----
  layernorm_f16<<<MTOK_, 256, 0, stream>>>(hbuf, norm_g, norm_b, lnf);
  meanpool<<<cdiv_l((long)B_ * E_, 256), 256, 0, stream>>>(lnf, pooled);
  {
    dim3 g((CLS_ + 127) / 128, 1);  // M=128 exactly one tile row
    gemm_wmma<0, true><<<g, 256, 0, stream>>>(pooled, wh, head_b, nullptr,
                                              nullptr, out, nullptr, B_, CLS_,
                                              E_);
  }
}
